// MPNN_LSTM_21002390077747
// MI455X (gfx1250) — matrix operand order repaired and checked by reading, not compiled, not run on hardware
//
#include <hip/hip_runtime.h>
#include <cstdint>
#include <cstddef>

// ---------------------------------------------------------------------------
// MPNN-LSTM forward for MI455X (gfx1250, wave32, WMMA).
// All GEMMs run on v_wmma_f32_16x16x32_bf16 (fp32 accumulate).
// GCN aggregation double-buffers its B-operand slab in LDS using
// global_load_async_to_lds_b128 (ASYNCcnt) pipelined with s_wait_asynccnt 1,
// overlapping the global->LDS fill of slab k+1 with the WMMAs of slab k.
// ---------------------------------------------------------------------------

typedef unsigned short u16;
typedef __attribute__((ext_vector_type(16))) __bf16 v16bf;
typedef __attribute__((ext_vector_type(8)))  float  v8f;

#define B_    8
#define T_    16
#define N_    512
#define F_IN  16
#define H_    64
#define M_    128            // B*T diagonal blocks
#define ROWS  (M_*N_)        // 65536
#define BN_   (B_*N_)        // 4096 LSTM batch

__device__ __forceinline__ u16 f2bf(float x) {
  unsigned u = __float_as_uint(x);
  unsigned r = (u + 0x7FFFu + ((u >> 16) & 1u)) >> 16;   // RNE truncation
  return (u16)r;
}

union VA { v16bf v; u16 u[16]; };

// A-operand (16x32 bf16): lane row = lane%16, k0 = (lane/16)*8.
// elems 0..7 -> K = k0..k0+7 ; elems 8..15 -> K = 16+k0 .. 16+k0+7
__device__ __forceinline__ v16bf load_a16(const u16* p /* &M[row][kbase] */, int half) {
  VA r;
  *(uint4*)&r.u[0] = *(const uint4*)(p + half * 8);
  *(uint4*)&r.u[8] = *(const uint4*)(p + 16 + half * 8);
  return r.v;
}

// B-operand (32x16 bf16) from a transposed matrix Bt[n][k]:
// lane col = lane%16, k0 = (lane/16)*8; elems 0..7 -> K = k0..k0+7, elems 8..15 -> K = 16+k0..16+k0+7.
__device__ __forceinline__ v16bf load_b16(const u16* p /* &Bt[col][kbase] */, int half) {
  VA r;
  *(uint4*)&r.u[0] = *(const uint4*)(p + half * 8);
  *(uint4*)&r.u[8] = *(const uint4*)(p + 16 + half * 8);
  return r.v;
}

__device__ __forceinline__ v8f wmma_bf16(v16bf a, v16bf b, v8f c) {
  return __builtin_amdgcn_wmma_f32_16x16x32_bf16(false, a, false, b, (short)0, c,
                                                 false, false);
}

__device__ __forceinline__ v8f v8f_zero() {
  v8f z = {0.f,0.f,0.f,0.f,0.f,0.f,0.f,0.f};
  return z;
}

// Async global -> LDS copy of 16 bytes per lane (tracked by ASYNCcnt).
// VDST supplies the wave-relative LDS byte address (generic LDS ptr low 32b).
__device__ __forceinline__ void async_copy_b128(u16* lds_dst, const u16* gsrc) {
  unsigned ldsoff = (unsigned)(size_t)(void*)lds_dst;
  asm volatile("global_load_async_to_lds_b128 %0, %1, off"
               :
               : "v"(ldsoff), "v"(gsrc)
               : "memory");
}
__device__ __forceinline__ void wait_async0() {   // all async copies done
  asm volatile("s_wait_asynccnt 0x0" ::: "memory");
}
__device__ __forceinline__ void wait_async1() {   // all but newest done (in-order)
  asm volatile("s_wait_asynccnt 0x1" ::: "memory");
}

// ---------------------------------------------------------------------------
// 1) dinv[m][c] = rsqrt(1 + sum_r A[m][r][c])     (self-loop adds the 1)
// ---------------------------------------------------------------------------
__global__ void k_dinv(const float* __restrict__ A, float* __restrict__ dinv) {
  int m = blockIdx.x;
  int c = blockIdx.y * 256 + threadIdx.x;
  const float* Am = A + (size_t)m * N_ * N_;
  float s = 1.0f;
  for (int r = 0; r < N_; ++r) s += Am[(size_t)r * N_ + c];
  dinv[m * N_ + c] = rsqrtf(s);
}

// ---------------------------------------------------------------------------
// 2) S[m][c][r] = bf16( (A[m][r][c] + (r==c)) * dinv[c] * dinv[r] )  (= norm^T)
//    64x64 tile transpose through LDS so both global read & write coalesce.
// ---------------------------------------------------------------------------
__global__ void k_buildS(const float* __restrict__ A, const float* __restrict__ dinv,
                         u16* __restrict__ S) {
  __shared__ float tile[64 * 65];
  int m  = blockIdx.x;
  int tr = blockIdx.y >> 3, tc = blockIdx.y & 7;
  int r0 = tr * 64, c0 = tc * 64;
  const float* Am = A + (size_t)m * N_ * N_;
  for (int i = 0; i < 16; ++i) {
    int e  = i * 256 + threadIdx.x;
    int rr = e >> 6, cc = e & 63;
    tile[rr * 65 + cc] = Am[(size_t)(r0 + rr) * N_ + c0 + cc];
  }
  __syncthreads();
  const float* dv = dinv + m * N_;
  for (int i = 0; i < 16; ++i) {
    int e  = i * 256 + threadIdx.x;
    int cl = e >> 6, rl = e & 63;
    int r = r0 + rl, c = c0 + cl;
    float v = tile[rl * 65 + cl] + ((r == c) ? 1.0f : 0.0f);
    S[((size_t)m * N_ + c) * N_ + r] = f2bf(v * dv[c] * dv[r]);
  }
}

// ---------------------------------------------------------------------------
// 3) Pt1[m][h][r] = bf16( X[m][r][:] . W1[:,h] )   (K=16, stored TRANSPOSED
//    so the aggregation B-operand is contiguous-K). idx enumerates (m,h,r)
//    with r fastest -> coalesced stores.
// ---------------------------------------------------------------------------
__global__ void k_xw1T(const float* __restrict__ X, const float* __restrict__ W1,
                       u16* __restrict__ Pt1) {
  int idx = blockIdx.x * 256 + threadIdx.x;   // M*H*N
  int r  = idx & 511;
  int mh = idx >> 9;
  int h  = mh & 63;
  int m  = mh >> 6;
  const float* xr = X + ((size_t)m * N_ + r) * F_IN;
  float acc = 0.f;
#pragma unroll
  for (int k = 0; k < F_IN; ++k) acc += xr[k] * W1[k * H_ + h];
  Pt1[idx] = f2bf(acc);
}

// ---------------------------------------------------------------------------
// 4) GCN aggregation: out[m][c][h] = relu( sum_r S[m][c][r]*Pt[m][h][r] + b[h] )
//    grid (M, 4), 256 threads (8 waves). Wave = 16 rows x 64 cols (4 tiles).
//    Pt k-slabs (64h x 32k bf16 = 4KB) double-buffered in LDS via async b128
//    copies; s_wait_asynccnt 1 overlaps slab k+1 fill with slab k WMMAs.
// ---------------------------------------------------------------------------
__device__ __forceinline__ void agg_slab_copy(u16* buf, const u16* Ptm, int kb,
                                              int ch_h, int ch_q) {
  async_copy_b128(buf + ch_h * 32 + ch_q, Ptm + (size_t)ch_h * N_ + kb + ch_q);
}

__device__ __forceinline__ void agg_step(const u16* __restrict__ Sm,
                                         const u16* buf, int c0, int lr, int half,
                                         int kb, v8f acc[4]) {
  const u16* ap = Sm + (size_t)(c0 + lr) * N_ + kb;
  if (kb + 32 < N_) __builtin_prefetch(ap + 32);   // global_prefetch_b8
  v16bf a = load_a16(ap, half);
#pragma unroll
  for (int tn = 0; tn < 4; ++tn) {
    v16bf b = load_b16(buf + (tn * 16 + lr) * 32, half);
    acc[tn] = wmma_bf16(a, b, acc[tn]);
  }
}

__global__ void k_gcn_agg(const u16* __restrict__ S, const u16* __restrict__ Pt,
                          const float* __restrict__ bias, float* __restrict__ out) {
  __shared__ __align__(16) u16 PtL0[64 * 32];   // slab buffer 0
  __shared__ __align__(16) u16 PtL1[64 * 32];   // slab buffer 1
  int m = blockIdx.x, strip = blockIdx.y;
  int tid  = threadIdx.x;
  int wid  = tid >> 5, lane = tid & 31;
  int half = lane >> 4, lr = lane & 15;
  int c0 = strip * 128 + wid * 16;
  const u16* Sm  = S  + (size_t)m * N_ * N_;
  const u16* Ptm = Pt + (size_t)m * H_ * N_;
  // chunk id for this thread: 256 chunks of 16B cover one 4KB slab
  int ch_h = tid >> 2, ch_q = (tid & 3) * 8;    // h row, k offset (ushorts)
  v8f acc[4];
  acc[0] = v8f_zero(); acc[1] = v8f_zero(); acc[2] = v8f_zero(); acc[3] = v8f_zero();

  agg_slab_copy(PtL0, Ptm, 0, ch_h, ch_q);                 // slab 0 in flight
#pragma unroll 1
  for (int ks = 0; ks < 14; ks += 2) {
    agg_slab_copy(PtL1, Ptm, (ks + 1) * 32, ch_h, ch_q);   // fill odd slab
    wait_async1();                                          // slab ks landed
    __syncthreads();
    agg_step(Sm, PtL0, c0, lr, half, ks * 32, acc);
    __syncthreads();
    agg_slab_copy(PtL0, Ptm, (ks + 2) * 32, ch_h, ch_q);   // fill next even slab
    wait_async1();                                          // slab ks+1 landed
    __syncthreads();
    agg_step(Sm, PtL1, c0, lr, half, (ks + 1) * 32, acc);
    __syncthreads();
  }
  // epilogue: slabs 14 (buf0, in flight) and 15 (buf1)
  agg_slab_copy(PtL1, Ptm, 15 * 32, ch_h, ch_q);
  wait_async1();
  __syncthreads();
  agg_step(Sm, PtL0, c0, lr, half, 14 * 32, acc);
  wait_async0();
  __syncthreads();
  agg_step(Sm, PtL1, c0, lr, half, 15 * 32, acc);

#pragma unroll
  for (int tn = 0; tn < 4; ++tn) {
    int col = tn * 16 + lr;
    float bv = bias[col];
#pragma unroll
    for (int v = 0; v < 8; ++v) {
      int row = c0 + 8 * half + v;
      float x = acc[tn][v] + bv;
      out[((size_t)m * N_ + row) * H_ + col] = x > 0.f ? x : 0.f;
    }
  }
}

// ---------------------------------------------------------------------------
// 5) BatchNorm (training-mode batch stats, biased var, eps=1e-5)
// ---------------------------------------------------------------------------
__global__ void k_bn_stats(const float* __restrict__ h, const float* __restrict__ g,
                           const float* __restrict__ be, float* __restrict__ scale,
                           float* __restrict__ shift) {
  __shared__ float s1[256], s2[256];
  int ch = blockIdx.x, tid = threadIdx.x;
  float a = 0.f, b = 0.f;
  for (int i = tid; i < ROWS; i += 256) {
    float x = h[(size_t)i * H_ + ch];
    a += x; b += x * x;
  }
  s1[tid] = a; s2[tid] = b; __syncthreads();
  for (int s = 128; s > 0; s >>= 1) {
    if (tid < s) { s1[tid] += s1[tid + s]; s2[tid] += s2[tid + s]; }
    __syncthreads();
  }
  if (tid == 0) {
    float mean = s1[0] / (float)ROWS;
    float var  = s2[0] / (float)ROWS - mean * mean;
    float sc = g[ch] * rsqrtf(var + 1e-5f);
    scale[ch] = sc; shift[ch] = be[ch] - mean * sc;
  }
}

__global__ void k_bn_apply(const float* __restrict__ h, const float* __restrict__ scale,
                           const float* __restrict__ shift, u16* __restrict__ out) {
  int idx = blockIdx.x * 256 + threadIdx.x;
  int ch = idx & 63;
  out[idx] = f2bf(h[idx] * scale[ch] + shift[ch]);
}

// ---------------------------------------------------------------------------
// 6) Generic Mx64 GEMM, A bf16 row-major, Bt bf16 [n][k]. One wave / 16 rows.
//    trans=1 stores output transposed per m-block: out[m][col][r] (for Pt2).
// ---------------------------------------------------------------------------
__global__ void k_gemm_bf(const u16* __restrict__ Abf, int lda,
                          const u16* __restrict__ Bt, int ldb, int K,
                          u16* __restrict__ out, int ldo, int trans) {
  int lane = threadIdx.x, half = lane >> 4, lr = lane & 15;
  int r0 = blockIdx.x * 16;
  v8f acc[4];
  acc[0] = v8f_zero(); acc[1] = v8f_zero(); acc[2] = v8f_zero(); acc[3] = v8f_zero();
  for (int kb = 0; kb < K; kb += 32) {
    v16bf a = load_a16(Abf + (size_t)(r0 + lr) * lda + kb, half);
#pragma unroll
    for (int tn = 0; tn < 4; ++tn) {
      v16bf b = load_b16(Bt + (size_t)(tn * 16 + lr) * ldb + kb, half);
      acc[tn] = wmma_bf16(a, b, acc[tn]);
    }
  }
#pragma unroll
  for (int tn = 0; tn < 4; ++tn) {
    int col = tn * 16 + lr;
#pragma unroll
    for (int v = 0; v < 8; ++v) {
      int rg = r0 + 8 * half + v;
      if (trans) {
        out[((size_t)(rg >> 9) * H_ + col) * N_ + (rg & 511)] = f2bf(acc[tn][v]);
      } else {
        out[(size_t)rg * ldo + col] = f2bf(acc[tn][v]);
      }
    }
  }
}

// ---------------------------------------------------------------------------
// 7) LSTM gate GEMM for one timestep:
//    gates[j][g] = sum_k x[j][k]*Wih[g][k] + sum_k h[j][k]*Whh[g][k]
//    x sources: layer1 -> (h1bn | h2bn) with time-major row remap; layer2 -> ys1[t].
//    grid (BN_/16, 4), 32 threads (one wave) per block.
// ---------------------------------------------------------------------------
__global__ void k_lstm_gates(const u16* __restrict__ x1, const u16* __restrict__ x2,
                             int seqmap, int t, int kin,
                             const u16* __restrict__ hbf,
                             const u16* __restrict__ Wih, const u16* __restrict__ Whh,
                             float* __restrict__ gates) {
  int lane = threadIdx.x, half = lane >> 4, lr = lane & 15;
  int jr0 = blockIdx.x * 16;
  int n0  = blockIdx.y * 64;
  int jr  = jr0 + lr;
  int rowx = seqmap ? (((jr >> 9) * T_ + t) * N_ + (jr & 511)) : jr;
  v8f acc[4];
  acc[0] = v8f_zero(); acc[1] = v8f_zero(); acc[2] = v8f_zero(); acc[3] = v8f_zero();
  int Ktot = kin + 64;
  for (int kb = 0; kb < Ktot; kb += 32) {
    if (kb < kin) {
      const u16* src = (kb < 64) ? x1 : x2;
      int koff = (kb < 64) ? kb : (kb - 64);
      v16bf a = load_a16(src + (size_t)rowx * 64 + koff, half);
#pragma unroll
      for (int tn = 0; tn < 4; ++tn) {
        v16bf b = load_b16(Wih + (size_t)(n0 + tn * 16 + lr) * kin + kb, half);
        acc[tn] = wmma_bf16(a, b, acc[tn]);
      }
    } else {
      v16bf a = load_a16(hbf + (size_t)jr * 64 + (kb - kin), half);
#pragma unroll
      for (int tn = 0; tn < 4; ++tn) {
        v16bf b = load_b16(Whh + (size_t)(n0 + tn * 16 + lr) * 64 + (kb - kin), half);
        acc[tn] = wmma_bf16(a, b, acc[tn]);
      }
    }
  }
#pragma unroll
  for (int tn = 0; tn < 4; ++tn) {
    int col = n0 + tn * 16 + lr;
#pragma unroll
    for (int v = 0; v < 8; ++v)
      gates[(size_t)(jr0 + 8 * half + v) * 256 + col] = acc[tn][v];
  }
}

// ---------------------------------------------------------------------------
// 8) LSTM cell update (torch gate order i,f,g,o). c-state fp32; h emitted bf16.
// ---------------------------------------------------------------------------
__global__ void k_lstm_cell(const float* __restrict__ gates,
                            const float* __restrict__ bih, const float* __restrict__ bhh,
                            float* __restrict__ cst, u16* __restrict__ hbf,
                            u16* __restrict__ ys, int t) {
  int idx = blockIdx.x * 256 + threadIdx.x;     // BN_*H
  int j = idx >> 6, hh = idx & 63;
  const float* gr = gates + (size_t)j * 256;
  float gi = gr[hh]        + bih[hh]        + bhh[hh];
  float gf = gr[64 + hh]   + bih[64 + hh]   + bhh[64 + hh];
  float gg = gr[128 + hh]  + bih[128 + hh]  + bhh[128 + hh];
  float go = gr[192 + hh]  + bih[192 + hh]  + bhh[192 + hh];
  float si = 1.f / (1.f + expf(-gi));
  float sf = 1.f / (1.f + expf(-gf));
  float so = 1.f / (1.f + expf(-go));
  float c = sf * cst[idx] + si * tanhf(gg);
  float h = so * tanhf(c);
  cst[idx] = c;
  u16 hb = f2bf(h);
  hbf[idx] = hb;
  if (ys) ys[(size_t)t * BN_ * H_ + idx] = hb;
}

// ---------------------------------------------------------------------------
// 9) skip path: skip[b*512+n][t*16+f] = X[b][t][n][f]  (bf16)
// ---------------------------------------------------------------------------
__global__ void k_skip(const float* __restrict__ X, u16* __restrict__ skipbf) {
  int idx = blockIdx.x * 256 + threadIdx.x;     // BN_*256
  int j = idx >> 8, col = idx & 255;
  int t = col >> 4, f = col & 15;
  int b = j >> 9, n = j & 511;
  skipbf[idx] = f2bf(X[(((size_t)(b * T_ + t)) * N_ + n) * F_IN + f]);
}

// ---------------------------------------------------------------------------
// 10) FC1: z1 = relu([hn1|hn2|skip] @ fc1W + b), K=384, N=64 via WMMA.
// ---------------------------------------------------------------------------
__global__ void k_fc1(const u16* __restrict__ hn1, const u16* __restrict__ hn2,
                      const u16* __restrict__ skip, const u16* __restrict__ Wt,
                      const float* __restrict__ bias, float* __restrict__ z1) {
  int lane = threadIdx.x, half = lane >> 4, lr = lane & 15;
  int r0 = blockIdx.x * 16;
  int jr = r0 + lr;
  v8f acc[4];
  acc[0] = v8f_zero(); acc[1] = v8f_zero(); acc[2] = v8f_zero(); acc[3] = v8f_zero();
  for (int kb = 0; kb < 384; kb += 32) {
    const u16* src; int ld, koff;
    if (kb < 64)       { src = hn1;  ld = 64;  koff = kb; }
    else if (kb < 128) { src = hn2;  ld = 64;  koff = kb - 64; }
    else               { src = skip; ld = 256; koff = kb - 128; }
    v16bf a = load_a16(src + (size_t)jr * ld + koff, half);
#pragma unroll
    for (int tn = 0; tn < 4; ++tn) {
      v16bf b = load_b16(Wt + (size_t)(tn * 16 + lr) * 384 + kb, half);
      acc[tn] = wmma_bf16(a, b, acc[tn]);
    }
  }
#pragma unroll
  for (int tn = 0; tn < 4; ++tn) {
    int col = tn * 16 + lr;
    float bv = bias[col];
#pragma unroll
    for (int v = 0; v < 8; ++v) {
      float x = acc[tn][v] + bv;
      z1[(size_t)(r0 + 8 * half + v) * 64 + col] = x > 0.f ? x : 0.f;
    }
  }
}

// 11) FC2: out[j] = relu(z1[j] . fc2W + b)   (K=64 -> OUT=1)
__global__ void k_fc2(const float* __restrict__ z1, const float* __restrict__ W,
                      const float* __restrict__ b, float* __restrict__ out) {
  int j = blockIdx.x * 256 + threadIdx.x;       // BN_
  float acc = b[0];
#pragma unroll
  for (int h = 0; h < 64; ++h) acc += z1[(size_t)j * 64 + h] * W[h];
  out[j] = acc > 0.f ? acc : 0.f;
}

// ----- small helpers -----
__global__ void k_conv(const float* __restrict__ in, u16* __restrict__ out, int n) {
  int i = blockIdx.x * 256 + threadIdx.x;
  if (i < n) out[i] = f2bf(in[i]);
}
__global__ void k_convT(const float* __restrict__ in, u16* __restrict__ out,
                        int K, int Nn, int total) {   // out[n][k] = in[k][n]
  int i = blockIdx.x * 256 + threadIdx.x;
  if (i < total) {
    int n = i / K, k = i - n * K;
    out[(size_t)n * K + k] = f2bf(in[(size_t)k * Nn + n]);
  }
}
__global__ void k_zero(unsigned* __restrict__ p, int n) {
  int i = blockIdx.x * 256 + threadIdx.x;
  if (i < n) p[i] = 0u;
}

// ---------------------------------------------------------------------------
extern "C" void kernel_launch(void* const* d_in, const int* in_sizes, int n_in,
                              void* d_out, int out_size, void* d_ws, size_t ws_size,
                              hipStream_t stream) {
  (void)in_sizes; (void)n_in; (void)out_size; (void)ws_size;

  const float* X    = (const float*)d_in[0];
  const float* A    = (const float*)d_in[2];
  const float* W1   = (const float*)d_in[3];
  const float* b1   = (const float*)d_in[4];
  const float* g1   = (const float*)d_in[5];
  const float* be1  = (const float*)d_in[6];
  const float* W2   = (const float*)d_in[7];
  const float* b2   = (const float*)d_in[8];
  const float* g2   = (const float*)d_in[9];
  const float* be2  = (const float*)d_in[10];
  const float* Wih1 = (const float*)d_in[11];
  const float* Whh1 = (const float*)d_in[12];
  const float* bih1 = (const float*)d_in[13];
  const float* bhh1 = (const float*)d_in[14];
  const float* Wih2 = (const float*)d_in[15];
  const float* Whh2 = (const float*)d_in[16];
  const float* bih2 = (const float*)d_in[17];
  const float* bhh2 = (const float*)d_in[18];
  const float* fc1W = (const float*)d_in[19];
  const float* fc1b = (const float*)d_in[20];
  const float* fc2W = (const float*)d_in[21];
  const float* fc2b = (const float*)d_in[22];
  float* out = (float*)d_out;

  // ---- workspace layout (~154 MB, each buffer 256B aligned) ----
  uint8_t* w = (uint8_t*)d_ws;
  size_t off = 0;
  auto take = [&](size_t bytes) -> uint8_t* {
    uint8_t* p = w + off;
    off = (off + bytes + 255) & ~(size_t)255;
    return p;
  };
  u16*   S      = (u16*)  take((size_t)M_ * N_ * N_ * 2);   // norm^T bf16 (64 MiB)
  float* dinv   = (float*)take((size_t)M_ * N_ * 4);
  u16*   Pt1    = (u16*)  take((size_t)ROWS * H_ * 2);      // [m][h][r]
  float* h1f    = (float*)take((size_t)ROWS * H_ * 4);
  u16*   h1bn   = (u16*)  take((size_t)ROWS * H_ * 2);
  u16*   Pt2    = (u16*)  take((size_t)ROWS * H_ * 2);      // [m][h][r]
  float* h2f    = (float*)take((size_t)ROWS * H_ * 4);
  u16*   h2bn   = (u16*)  take((size_t)ROWS * H_ * 2);
  float* scale1 = (float*)take(64 * 4);
  float* shift1 = (float*)take(64 * 4);
  float* scale2 = (float*)take(64 * 4);
  float* shift2 = (float*)take(64 * 4);
  u16*   W2t    = (u16*)  take(64 * 64 * 2);
  u16*   Wih1b  = (u16*)  take(256 * 128 * 2);
  u16*   Whh1b  = (u16*)  take(256 * 64 * 2);
  u16*   Wih2b  = (u16*)  take(256 * 64 * 2);
  u16*   Whh2b  = (u16*)  take(256 * 64 * 2);
  u16*   fc1Wt  = (u16*)  take(64 * 384 * 2);
  float* gates  = (float*)take((size_t)BN_ * 256 * 4);
  u16*   hL1    = (u16*)  take((size_t)BN_ * H_ * 2);       // contiguous zero region
  float* cL1    = (float*)take((size_t)BN_ * H_ * 4);
  u16*   hL2    = (u16*)  take((size_t)BN_ * H_ * 2);
  float* cL2    = (float*)take((size_t)BN_ * H_ * 4);
  u16*   ys1    = (u16*)  take((size_t)T_ * BN_ * H_ * 2);
  u16*   skipbf = (u16*)  take((size_t)BN_ * 256 * 2);
  float* z1     = (float*)take((size_t)BN_ * H_ * 4);

  // ---- graph normalization ----
  k_dinv  <<<dim3(M_, 2),  256, 0, stream>>>(A, dinv);
  k_buildS<<<dim3(M_, 64), 256, 0, stream>>>(A, dinv, S);

  // ---- weight conversions ----
  k_convT<<<(64 * 64 + 255) / 256, 256, 0, stream>>>(W2, W2t, 64, 64, 64 * 64);
  k_conv <<<(256 * 128 + 255) / 256, 256, 0, stream>>>(Wih1, Wih1b, 256 * 128);
  k_conv <<<(256 * 64 + 255) / 256, 256, 0, stream>>>(Whh1, Whh1b, 256 * 64);
  k_conv <<<(256 * 64 + 255) / 256, 256, 0, stream>>>(Wih2, Wih2b, 256 * 64);
  k_conv <<<(256 * 64 + 255) / 256, 256, 0, stream>>>(Whh2, Whh2b, 256 * 64);
  k_convT<<<(384 * 64 + 255) / 256, 256, 0, stream>>>(fc1W, fc1Wt, 384, 64, 384 * 64);

  // ---- GCN layer 1 ----
  k_xw1T   <<<ROWS * H_ / 256, 256, 0, stream>>>(X, W1, Pt1);
  k_gcn_agg<<<dim3(M_, 4), 256, 0, stream>>>(S, Pt1, b1, h1f);
  k_bn_stats<<<64, 256, 0, stream>>>(h1f, g1, be1, scale1, shift1);
  k_bn_apply<<<ROWS * H_ / 256, 256, 0, stream>>>(h1f, scale1, shift1, h1bn);

  // ---- GCN layer 2 ----
  k_gemm_bf<<<ROWS / 16, 32, 0, stream>>>(h1bn, H_, W2t, 64, 64, Pt2, H_, /*trans=*/1);
  k_gcn_agg<<<dim3(M_, 4), 256, 0, stream>>>(S, Pt2, b2, h2f);
  k_bn_stats<<<64, 256, 0, stream>>>(h2f, g2, be2, scale2, shift2);
  k_bn_apply<<<ROWS * H_ / 256, 256, 0, stream>>>(h2f, scale2, shift2, h2bn);

  // ---- skip path + LSTM state init ----
  k_skip<<<BN_ * 256 / 256, 256, 0, stream>>>(X, skipbf);
  {
    size_t zero_bytes = (size_t)BN_ * H_ * 2 + (size_t)BN_ * H_ * 4 +
                        (size_t)BN_ * H_ * 2 + (size_t)BN_ * H_ * 4;   // hL1,cL1,hL2,cL2
    int n32 = (int)(zero_bytes / 4);
    k_zero<<<(n32 + 255) / 256, 256, 0, stream>>>((unsigned*)hL1, n32);
  }

  // ---- LSTM layer 1 (input 2H = [h1bn|h2bn], time-major row remap) ----
  for (int t = 0; t < T_; ++t) {
    k_lstm_gates<<<dim3(BN_ / 16, 4), 32, 0, stream>>>(
        h1bn, h2bn, /*seqmap=*/1, t, /*kin=*/128, hL1, Wih1b, Whh1b, gates);
    k_lstm_cell<<<BN_ * H_ / 256, 256, 0, stream>>>(
        gates, bih1, bhh1, cL1, hL1, ys1, t);
  }
  // ---- LSTM layer 2 (input = ys1[t]) ----
  for (int t = 0; t < T_; ++t) {
    k_lstm_gates<<<dim3(BN_ / 16, 4), 32, 0, stream>>>(
        ys1 + (size_t)t * BN_ * H_, nullptr, /*seqmap=*/0, t, /*kin=*/64,
        hL2, Wih2b, Whh2b, gates);
    k_lstm_cell<<<BN_ * H_ / 256, 256, 0, stream>>>(
        gates, bih2, bhh2, cL2, hL2, nullptr, t);
  }

  // ---- FC heads ----
  k_fc1<<<BN_ / 16, 32, 0, stream>>>(hL1, hL2, skipbf, fc1Wt, fc1b, z1);
  k_fc2<<<BN_ / 256, 256, 0, stream>>>(z1, fc2W, fc2b, out);
}
